// ReservoirLayer_90314572300891
// MI455X (gfx1250) — compile-verified, matrix operand order checked by reading, not run on hardware
//
#include <hip/hip_runtime.h>
#include <hip/hip_bf16.h>
#include <stdint.h>

// Problem shape (compile-time constants, match the reference).
#define B_   32
#define T_   512
#define DIN  256
#define DOUT 256
#define R_   2048

typedef __attribute__((ext_vector_type(16))) __bf16    v16bf;
typedef __attribute__((ext_vector_type(8)))  float     v8f;
typedef __attribute__((ext_vector_type(4)))  uint32_t  u32x4;

struct BF16Frag { u32x4 lo, hi; };  // 32 bytes = 16 bf16

// Load one wave-lane's share of a 16x32 bf16 WMMA operand.
// Per CDNA5 ISA (CDNA5 ISA, 16-bit A 16x32): lane L<16 -> row M=L holds
// K = k..k+7 (v0..3) and K = k+16..k+23 (v4..7); lane L>=16 -> same row,
// K shifted by +8.  Caller bakes (row * stride + khalf*8) into `p`.
__device__ __forceinline__ v16bf load_frag16(const uint16_t* p) {
  BF16Frag f;
  f.lo = *(const u32x4*)(p);        // K+0 .. K+7   (16B aligned)
  f.hi = *(const u32x4*)(p + 16);   // K+16 .. K+23 (16B aligned)
  return __builtin_bit_cast(v16bf, f);
}

__device__ __forceinline__ v8f wmma_bf16(v16bf a, v16bf b, v8f c) {
  return __builtin_amdgcn_wmma_f32_16x16x32_bf16(false, a, false, b,
                                                 (short)0, c, false, false);
}

__device__ __forceinline__ uint16_t f32_to_bf16_rne(float v) {
  uint32_t u = __float_as_uint(v);
  return (uint16_t)((u + 0x7FFFu + ((u >> 16) & 1u)) >> 16);
}

// ---------------------------------------------------------------------------
// One-time fp32 -> bf16 conversion (weights / inputs), grid-stride.
// ---------------------------------------------------------------------------
__global__ void f32_to_bf16_kernel(const float* __restrict__ in,
                                   uint16_t* __restrict__ out, int n) {
  int i = blockIdx.x * blockDim.x + threadIdx.x;
  int stride = gridDim.x * blockDim.x;
  for (; i < n; i += stride) out[i] = f32_to_bf16_rne(in[i]);
}

// ---------------------------------------------------------------------------
// One reservoir timestep with 2x2 register blocking:
//   acc = x_t @ W_in^T + state_old @ W_res^T            (bf16 WMMA, f32 acc)
//   state_new = (1-alpha)*state_old + alpha*tanh(acc)   (f32)
// Each wave computes a 32x32 tile (both 16-row batch halves x two 16-col
// tiles): 4 WMMAs per 8 b128 loads, 4 independent accumulator chains.
// Block = 128 threads (4 waves) covers 32x128; grid = (R/128, 1) = (16, 1).
// Double-buffered state: reads *_old, writes *_new (no intra-launch race).
// ---------------------------------------------------------------------------
__global__ void __launch_bounds__(128)
reservoir_step_kernel(const uint16_t* __restrict__ x_bf,      // [B][T][DIN]
                      const uint16_t* __restrict__ win_bf,    // [R][DIN]
                      const uint16_t* __restrict__ wres_bf,   // [R][R]
                      const float*    __restrict__ alpha_p,   // scalar
                      const float*    __restrict__ state_old_f32,  // [B][R]
                      const uint16_t* __restrict__ state_old_bf,   // [B][R]
                      float*          __restrict__ state_new_f32,  // [B][R]
                      uint16_t*       __restrict__ state_new_bf,   // [B][R]
                      int t) {
  const int lane  = threadIdx.x & 31;
  const int wave  = threadIdx.x >> 5;
  const int r0    = blockIdx.x * 128 + wave * 32;  // two col tiles: r0, r0+16
  const int l15   = lane & 15;
  const int khalf = lane >> 4;                     // K sub-offset select

  v8f acc00 = {}, acc01 = {}, acc10 = {}, acc11 = {};  // [rowTile][colTile]

  // ---- u_t = x_t @ W_in^T  (K = DIN = 256 -> 8 iters x 4 WMMAs) ----
  {
    const uint16_t* a0 = x_bf + ((size_t)(l15)      * T_ + t) * DIN + khalf * 8;
    const uint16_t* a1 = x_bf + ((size_t)(16 + l15) * T_ + t) * DIN + khalf * 8;
    const uint16_t* w0 = win_bf + (size_t)(r0 + l15)      * DIN + khalf * 8;
    const uint16_t* w1 = win_bf + (size_t)(r0 + 16 + l15) * DIN + khalf * 8;
#pragma unroll
    for (int k = 0; k < DIN; k += 32) {
      v16bf va0 = load_frag16(a0 + k);
      v16bf va1 = load_frag16(a1 + k);
      v16bf vb0 = load_frag16(w0 + k);
      v16bf vb1 = load_frag16(w1 + k);
      acc00 = wmma_bf16(va0, vb0, acc00);
      acc01 = wmma_bf16(va0, vb1, acc01);
      acc10 = wmma_bf16(va1, vb0, acc10);
      acc11 = wmma_bf16(va1, vb1, acc11);
    }
  }

  // ---- res = state_old @ W_res^T  (K = R = 2048 -> 64 iters x 4 WMMAs) ----
  {
    const uint16_t* a0 = state_old_bf + (size_t)(l15)      * R_ + khalf * 8;
    const uint16_t* a1 = state_old_bf + (size_t)(16 + l15) * R_ + khalf * 8;
    const uint16_t* w0 = wres_bf + (size_t)(r0 + l15)      * R_ + khalf * 8;
    const uint16_t* w1 = wres_bf + (size_t)(r0 + 16 + l15) * R_ + khalf * 8;
    for (int k = 0; k < R_; k += 32) {
      __builtin_prefetch(w0 + k + 256, 0, 0);   // global_prefetch_b8, 8 iters ahead
      __builtin_prefetch(w1 + k + 256, 0, 0);
      v16bf va0 = load_frag16(a0 + k);
      v16bf va1 = load_frag16(a1 + k);
      v16bf vb0 = load_frag16(w0 + k);
      v16bf vb1 = load_frag16(w1 + k);
      acc00 = wmma_bf16(va0, vb0, acc00);
      acc01 = wmma_bf16(va0, vb1, acc01);
      acc10 = wmma_bf16(va1, vb0, acc10);
      acc11 = wmma_bf16(va1, vb1, acc11);
    }
  }

  // ---- leaky tanh update on the four C fragments ----
  // C/D layout per tile: lane = n + 16*(m>=8); VGPR j -> m = j + 8*khalf.
  const float alpha = *alpha_p;
  const float beta  = 1.0f - alpha;
  const v8f accs[4] = {acc00, acc01, acc10, acc11};
#pragma unroll
  for (int rt = 0; rt < 2; ++rt) {
#pragma unroll
    for (int ct = 0; ct < 2; ++ct) {
      const v8f a = accs[rt * 2 + ct];
      const int bbase = rt * 16;
      const int col   = r0 + ct * 16 + l15;
#pragma unroll
      for (int j = 0; j < 8; ++j) {
        const int m = j + 8 * khalf;
        const size_t idx = (size_t)(bbase + m) * R_ + (size_t)col;
        const float nv = beta * state_old_f32[idx] + alpha * tanhf(a[j]);
        state_new_f32[idx] = nv;
        state_new_bf[idx]  = f32_to_bf16_rne(nv);
      }
    }
  }
}

// ---------------------------------------------------------------------------
// out_t = state_new @ W_out^T + b_out   -> out[b][t][:]
// Same 2x2 register blocking: each wave computes 32 rows x 32 cols.
// Block = 128 threads (4 waves); grid = (DOUT/128, 1) = (2, 1).
// ---------------------------------------------------------------------------
__global__ void __launch_bounds__(128)
reservoir_out_kernel(const uint16_t* __restrict__ state_bf,  // [B][R]
                     const uint16_t* __restrict__ wout_bf,   // [DOUT][R]
                     const float*    __restrict__ b_out,     // [DOUT]
                     float*          __restrict__ out,       // [B][T][DOUT]
                     int t) {
  const int lane  = threadIdx.x & 31;
  const int wave  = threadIdx.x >> 5;
  const int o0    = blockIdx.x * 128 + wave * 32;
  const int l15   = lane & 15;
  const int khalf = lane >> 4;

  v8f acc00 = {}, acc01 = {}, acc10 = {}, acc11 = {};

  const uint16_t* a0 = state_bf + (size_t)(l15)      * R_ + khalf * 8;
  const uint16_t* a1 = state_bf + (size_t)(16 + l15) * R_ + khalf * 8;
  const uint16_t* w0 = wout_bf + (size_t)(o0 + l15)      * R_ + khalf * 8;
  const uint16_t* w1 = wout_bf + (size_t)(o0 + 16 + l15) * R_ + khalf * 8;
  for (int k = 0; k < R_; k += 32) {
    __builtin_prefetch(w0 + k + 256, 0, 0);
    __builtin_prefetch(w1 + k + 256, 0, 0);
    v16bf va0 = load_frag16(a0 + k);
    v16bf va1 = load_frag16(a1 + k);
    v16bf vb0 = load_frag16(w0 + k);
    v16bf vb1 = load_frag16(w1 + k);
    acc00 = wmma_bf16(va0, vb0, acc00);
    acc01 = wmma_bf16(va0, vb1, acc01);
    acc10 = wmma_bf16(va1, vb0, acc10);
    acc11 = wmma_bf16(va1, vb1, acc11);
  }

  const v8f accs[4] = {acc00, acc01, acc10, acc11};
#pragma unroll
  for (int rt = 0; rt < 2; ++rt) {
#pragma unroll
    for (int ct = 0; ct < 2; ++ct) {
      const v8f a = accs[rt * 2 + ct];
      const int bbase = rt * 16;
      const int col   = o0 + ct * 16 + l15;
      const float bias = b_out[col];
#pragma unroll
      for (int j = 0; j < 8; ++j) {
        const int m = j + 8 * khalf;
        out[((size_t)(bbase + m) * T_ + t) * DOUT + (size_t)col] = a[j] + bias;
      }
    }
  }
}

// ---------------------------------------------------------------------------
// Host-side launch: convert to bf16 once, then 512 (step, out) kernel pairs.
// All stream-ordered -> graph-capturable; cross-step dependency is handled
// by stream serialization, W_res (8 MB bf16) stays resident in the 192MB L2.
// ---------------------------------------------------------------------------
extern "C" void kernel_launch(void* const* d_in, const int* in_sizes, int n_in,
                              void* d_out, int out_size, void* d_ws, size_t ws_size,
                              hipStream_t stream) {
  (void)in_sizes; (void)n_in; (void)out_size; (void)ws_size;

  const float* x     = (const float*)d_in[0];  // [B][T][DIN]
  const float* W_in  = (const float*)d_in[1];  // [R][DIN]
  const float* W_res = (const float*)d_in[2];  // [R][R]
  const float* W_out = (const float*)d_in[3];  // [DOUT][R]
  const float* b_out = (const float*)d_in[4];  // [DOUT]
  const float* alpha = (const float*)d_in[5];  // scalar
  float* out = (float*)d_out;

  // Workspace carve-up (all offsets 256B-aligned by construction).
  const size_t NX    = (size_t)B_ * T_ * DIN;  // 4,194,304
  const size_t NRES  = (size_t)R_ * R_;        // 4,194,304
  const size_t NWIN  = (size_t)R_ * DIN;       //   524,288
  const size_t NWOUT = (size_t)DOUT * R_;      //   524,288
  const size_t NST   = (size_t)B_ * R_;        //    65,536

  char* ws = (char*)d_ws;
  size_t off = 0;
  uint16_t* x_bf    = (uint16_t*)(ws + off); off += NX    * 2;
  uint16_t* wres_bf = (uint16_t*)(ws + off); off += NRES  * 2;
  uint16_t* win_bf  = (uint16_t*)(ws + off); off += NWIN  * 2;
  uint16_t* wout_bf = (uint16_t*)(ws + off); off += NWOUT * 2;
  float* sf[2];
  sf[0] = (float*)(ws + off); off += NST * 4;
  sf[1] = (float*)(ws + off); off += NST * 4;
  uint16_t* sb[2];
  sb[0] = (uint16_t*)(ws + off); off += NST * 2;
  sb[1] = (uint16_t*)(ws + off); off += NST * 2;

  // One-time conversions + zero-init of state double buffers.
  f32_to_bf16_kernel<<<2048, 256, 0, stream>>>(x,     x_bf,    (int)NX);
  f32_to_bf16_kernel<<<2048, 256, 0, stream>>>(W_res, wres_bf, (int)NRES);
  f32_to_bf16_kernel<<<512,  256, 0, stream>>>(W_in,  win_bf,  (int)NWIN);
  f32_to_bf16_kernel<<<512,  256, 0, stream>>>(W_out, wout_bf, (int)NWOUT);
  hipMemsetAsync(sf[0], 0, NST * 4, stream);
  hipMemsetAsync(sb[0], 0, NST * 2, stream);

  const dim3 stepGrid(R_ / 128, 1);    // (16, 1): 64 waves, 32x128 per block
  const dim3 outGrid(DOUT / 128, 1);   // (2, 1)

  for (int t = 0; t < T_; ++t) {
    const int cur = t & 1;             // old buffer index
    const int nxt = cur ^ 1;           // new buffer index
    reservoir_step_kernel<<<stepGrid, 128, 0, stream>>>(
        x_bf, win_bf, wres_bf, alpha,
        sf[cur], sb[cur], sf[nxt], sb[nxt], t);
    reservoir_out_kernel<<<outGrid, 128, 0, stream>>>(
        sb[nxt], wout_bf, b_out, out, t);
  }
}
